// ConsisGAD_46377056862940
// MI455X (gfx1250) — matrix operand order repaired and checked by reading, not compiled, not run on hardware
//
#include <hip/hip_runtime.h>
#include <hip/hip_bf16.h>

typedef __attribute__((ext_vector_type(16))) _Float16 v16h;
typedef __attribute__((ext_vector_type(8)))  _Float16 v8h;
typedef __attribute__((ext_vector_type(8)))  float    v8f;

#define N_NODES 50000
#define N_EDGES 800000
#define HID 128
#define LDS_STRIDE 136   // 16 rows * 136 halves, padded to dodge bank conflicts

// ---------- WMMA fragment helpers (layouts per CDNA5 ISA 7.12.2) ----------

// A fragment (16x32, f16): lane holds row M=lane%16.
// lanes 0-15 : elems 0..7 -> K=kb+0..7,   elems 8..15 -> K=kb+16..23
// lanes 16-31: elems 0..7 -> K=kb+8..15,  elems 8..15 -> K=kb+24..31
template <typename HP>
__device__ __forceinline__ v16h load_a_frag(const HP* rowp, int kb, int khalf) {
    const HP* p0 = rowp + kb + khalf * 8;
    const HP* p1 = rowp + kb + 16 + khalf * 8;
    v16h a;
#pragma unroll
    for (int i = 0; i < 8; ++i) {
        a[i]     = p0[i];
        a[i + 8] = p1[i];
    }
    return a;
}

// A fragment built from an fp32 row (converted on the fly)
__device__ __forceinline__ v16h load_a_frag_f32(const float* rowp, int kb, int khalf) {
    const float* p0 = rowp + kb + khalf * 8;
    const float* p1 = rowp + kb + 16 + khalf * 8;
    v16h a;
#pragma unroll
    for (int i = 0; i < 8; ++i) {
        a[i]     = (_Float16)p0[i];
        a[i + 8] = (_Float16)p1[i];
    }
    return a;
}

// B fragment (32x16, f16) from W stored transposed as Wt[n][k] (ld = K dim):
// lane holds col N=lane%16; elems 0..7 -> K = kb + khalf*8 + i, elems 8..15 -> K = kb + 16 + khalf*8 + (i-8)
__device__ __forceinline__ v16h load_b_frag(const _Float16* Wt, int ld, int n, int kb, int khalf) {
    const _Float16* p = Wt + (size_t)n * ld + kb + khalf * 8;
    v16h b;
#pragma unroll
    for (int i = 0; i < 8; ++i) {
        b[i]     = p[i];
        b[i + 8] = p[i + 16];
    }
    return b;
}

// ---------- prep kernels ----------

// Wt[n*K + k] = (f16) W[k*Ncols + n]
__global__ void cvt_transpose_kernel(const float* __restrict__ W, _Float16* __restrict__ Wt,
                                     int K, int Ncols) {
    int idx = blockIdx.x * blockDim.x + threadIdx.x;
    if (idx >= K * Ncols) return;
    int k = idx / Ncols;
    int n = idx % Ncols;
    Wt[(size_t)n * K + k] = (_Float16)W[idx];
}

__global__ void zero_f32_kernel(float* __restrict__ p, int n) {
    int i = blockIdx.x * blockDim.x + threadIdx.x;
    if (i < n) p[i] = 0.0f;
}

// ---------- kernel 1: h = relu(x @ W_in + b_in), stored f16 ----------

__global__ void proj_kernel(const float* __restrict__ x, const _Float16* __restrict__ WinT,
                            const float* __restrict__ b_in, _Float16* __restrict__ h) {
    const int lane  = threadIdx.x & 31;
    const int wave  = threadIdx.x >> 5;
    const int tile  = blockIdx.x * 8 + wave;
    if (tile * 16 >= N_NODES) return;
    const int khalf = lane >> 4;
    const int lm    = lane & 15;

    const float* xrow = x + (size_t)(tile * 16 + lm) * HID;

    v8f acc[8] = {};
#pragma unroll
    for (int kb = 0; kb < HID; kb += 32) {
        v16h a = load_a_frag_f32(xrow, kb, khalf);
#pragma unroll
        for (int nt = 0; nt < 8; ++nt) {
            v16h b = load_b_frag(WinT, HID, nt * 16 + lm, kb, khalf);
            acc[nt] = __builtin_amdgcn_wmma_f32_16x16x32_f16(
                false, a, false, b, (short)0, acc[nt], false, false);
        }
    }
#pragma unroll
    for (int nt = 0; nt < 8; ++nt) {
        const int n = nt * 16 + lm;
        const float bias = b_in[n];
#pragma unroll
        for (int r = 0; r < 8; ++r) {
            const int m = tile * 16 + r + 8 * khalf;
            float v = acc[nt][r] + bias;
            v = v > 0.0f ? v : 0.0f;
            h[(size_t)m * HID + n] = (_Float16)v;
        }
    }
}

// ---------- kernel 2: edge MLP + atomic scatter-add ----------

__global__ void edge_kernel(const _Float16* __restrict__ h, const long long* __restrict__ ei,
                            const _Float16* __restrict__ W1T, const float* __restrict__ b1,
                            const _Float16* __restrict__ W2T, const float* __restrict__ b2,
                            float* __restrict__ h2) {
    __shared__ _Float16 tbuf[8][16 * LDS_STRIDE];

    const int lane  = threadIdx.x & 31;
    const int wave  = threadIdx.x >> 5;
    const int tile  = blockIdx.x * 8 + wave;       // exactly 50000 tiles, no tail
    const int khalf = lane >> 4;
    const int lm    = lane & 15;

    // per-lane A-row gather pointers (lane's matrix row = edge tile*16 + lm)
    const int e = tile * 16 + lm;
    const long long srow = ei[e];                  // edge_index[0] = src (x_j)
    const long long drow = ei[N_EDGES + e];        // edge_index[1] = dst (x_i)
    const _Float16* hs = h + (size_t)srow * HID;
    const _Float16* hd = h + (size_t)drow * HID;

    // t = relu(h[dst] @ W1[:H] + h[src] @ W1[H:] + b1)
    v8f acc[8] = {};
#pragma unroll
    for (int kb = 0; kb < HID; kb += 32) {
        v16h ad = load_a_frag(hd, kb, khalf);
        v16h as = load_a_frag(hs, kb, khalf);
#pragma unroll
        for (int nt = 0; nt < 8; ++nt) {
            const int n = nt * 16 + lm;
            v16h bd = load_b_frag(W1T, 2 * HID, n, kb, khalf);          // W1 rows [0,128)
            acc[nt] = __builtin_amdgcn_wmma_f32_16x16x32_f16(
                false, ad, false, bd, (short)0, acc[nt], false, false);
            v16h bs = load_b_frag(W1T, 2 * HID, n, HID + kb, khalf);    // W1 rows [128,256)
            acc[nt] = __builtin_amdgcn_wmma_f32_16x16x32_f16(
                false, as, false, bs, (short)0, acc[nt], false, false);
        }
    }

    // bias + relu -> LDS (as f16) so it can be reloaded in A-fragment layout
    _Float16* tw = tbuf[wave];
#pragma unroll
    for (int nt = 0; nt < 8; ++nt) {
        const int n = nt * 16 + lm;
        const float bias = b1[n];
#pragma unroll
        for (int r = 0; r < 8; ++r) {
            const int m = r + 8 * khalf;
            float v = acc[nt][r] + bias;
            v = v > 0.0f ? v : 0.0f;
            tw[m * LDS_STRIDE + n] = (_Float16)v;
        }
    }
    __syncthreads();

    // m = t @ W2 + b2
    v8f acc2[8] = {};
    const _Float16* trow = tw + lm * LDS_STRIDE;
#pragma unroll
    for (int kb = 0; kb < HID; kb += 32) {
        v16h a = load_a_frag(trow, kb, khalf);
#pragma unroll
        for (int nt = 0; nt < 8; ++nt) {
            v16h b = load_b_frag(W2T, HID, nt * 16 + lm, kb, khalf);
            acc2[nt] = __builtin_amdgcn_wmma_f32_16x16x32_f16(
                false, a, false, b, (short)0, acc2[nt], false, false);
        }
    }

    // segment_sum: atomic scatter into h2[dst]
#pragma unroll
    for (int r = 0; r < 8; ++r) {
        const int m = r + 8 * khalf;
        const long long dr = ei[N_EDGES + tile * 16 + m];
        float* outp = h2 + (size_t)dr * HID;
#pragma unroll
        for (int nt = 0; nt < 8; ++nt) {
            const int n = nt * 16 + lm;
            unsafeAtomicAdd(outp + n, acc2[nt][r] + b2[n]);
        }
    }
}

// ---------- kernel 3: out = relu(h2) @ Wc + bc (fp32, tiny N-dim) ----------

__global__ void cls_kernel(const float* __restrict__ h2, const float* __restrict__ Wc,
                           const float* __restrict__ bc, float* __restrict__ out) {
    int i = blockIdx.x * blockDim.x + threadIdx.x;
    if (i >= N_NODES) return;
    const float* row = h2 + (size_t)i * HID;
    float a0 = bc[0], a1 = bc[1];
#pragma unroll 8
    for (int k = 0; k < HID; ++k) {
        float v = row[k];
        v = v > 0.0f ? v : 0.0f;
        a0 += v * Wc[k * 2 + 0];
        a1 += v * Wc[k * 2 + 1];
    }
    out[i * 2 + 0] = a0;
    out[i * 2 + 1] = a1;
}

// ---------- launcher ----------

extern "C" void kernel_launch(void* const* d_in, const int* in_sizes, int n_in,
                              void* d_out, int out_size, void* d_ws, size_t ws_size,
                              hipStream_t stream) {
    const float*     x    = (const float*)d_in[0];
    const long long* ei   = (const long long*)d_in[1];   // int64 [2][E]
    const float*     W_in = (const float*)d_in[2];
    const float*     b_in = (const float*)d_in[3];
    const float*     W1   = (const float*)d_in[4];
    const float*     b1   = (const float*)d_in[5];
    const float*     W2   = (const float*)d_in[6];
    const float*     b2   = (const float*)d_in[7];
    const float*     Wc   = (const float*)d_in[8];
    const float*     bc   = (const float*)d_in[9];
    float* out = (float*)d_out;

    char* ws = (char*)d_ws;
    // workspace layout (all 16B aligned)
    _Float16* h_f16 = (_Float16*)(ws);                               // 50000*128*2  = 12,800,000 B
    float*    h2    = (float*)(ws + 12800000);                       // 50000*128*4  = 25,600,000 B
    _Float16* WinT  = (_Float16*)(ws + 38400000);                    // 128*128*2    = 32,768 B
    _Float16* W1T   = (_Float16*)(ws + 38400000 + 32768);            // 256*128*2    = 65,536 B
    _Float16* W2T   = (_Float16*)(ws + 38400000 + 32768 + 65536);    // 128*128*2    = 32,768 B

    // weight conversion / transposition to f16 [n][k]
    cvt_transpose_kernel<<<(128 * 128 + 255) / 256, 256, 0, stream>>>(W_in, WinT, 128, 128);
    cvt_transpose_kernel<<<(256 * 128 + 255) / 256, 256, 0, stream>>>(W1, W1T, 256, 128);
    cvt_transpose_kernel<<<(128 * 128 + 255) / 256, 256, 0, stream>>>(W2, W2T, 128, 128);

    // zero the aggregation buffer
    zero_f32_kernel<<<(N_NODES * HID + 255) / 256, 256, 0, stream>>>(h2, N_NODES * HID);

    // h = relu(x @ W_in + b_in): 3125 tiles of 16 rows, 8 waves/block
    proj_kernel<<<(3125 + 7) / 8, 256, 0, stream>>>(x, WinT, b_in, h_f16);

    // edge MLP + scatter: 50000 tiles of 16 edges, 8 waves/block -> 6250 blocks
    edge_kernel<<<N_EDGES / 16 / 8, 256, 0, stream>>>(h_f16, ei, W1T, b1, W2T, b2, h2);

    // classifier head
    cls_kernel<<<(N_NODES + 255) / 256, 256, 0, stream>>>(h2, Wc, bc, out);
}
